// GATConv_22179211116710
// MI455X (gfx1250) — matrix operand order repaired and checked by reading, not compiled, not run on hardware
//
#include <hip/hip_runtime.h>
#include <hip/hip_bf16.h>
#include <math.h>

#define N_NODES   50000
#define N_EDGES   800000
#define IN_FEATS  512
#define NUM_HEADS 4
#define OUT_FEATS 64
#define HD        (NUM_HEADS * OUT_FEATS)   // 256
#define NEG_SLOPE 0.2f

typedef __attribute__((ext_vector_type(16))) __bf16        v16bf;
typedef __attribute__((ext_vector_type(8)))  float         v8f;
typedef __attribute__((ext_vector_type(4)))  unsigned int  v4u;
typedef __attribute__((ext_vector_type(8)))  int           v8i;
typedef __attribute__((ext_vector_type(4)))  int           v4i;

union Frag16 { v16bf v; unsigned int d[8]; unsigned short u[16]; };

__device__ __forceinline__ unsigned short f32_to_bf16(float f) {
    unsigned int u = __float_as_uint(f);
    unsigned int r = u + 0x7FFFu + ((u >> 16) & 1u);   // round-to-nearest-even
    return (unsigned short)(r >> 16);
}

__device__ __forceinline__ void atomicMaxFloat(float* addr, float val) {
    // signed/unsigned punning trick: correct total order for IEEE floats
    if (val >= 0.0f)
        atomicMax((int*)addr, __float_as_int(val));
    else
        atomicMin((unsigned int*)addr, __float_as_uint(val));
}

// ---------------------------------------------------------------- utilities
__global__ void fill_f32(float* p, float v, int n) {
    int i = blockIdx.x * blockDim.x + threadIdx.x;
    if (i < n) p[i] = v;
}

__global__ void cvt_w_bf16(const float* __restrict__ W, unsigned short* __restrict__ Wb, int n) {
    int i = blockIdx.x * blockDim.x + threadIdx.x;
    if (i < n) Wb[i] = f32_to_bf16(W[i]);
}

// ---------------------------------------------------------------- GEMM: ft = feat @ W  (bf16 WMMA, f32 acc)
// block = 256 threads (8 waves). Block b owns rows [16b, 16b+16). 50000 = 3125*16 exact.
// Wave w computes two 16x16 tiles at column bases 32w and 32w+16.
// B chunks (32x256 bf16 of W) are DMA'd into LDS by the Tensor Data Mover with
// pad_interval=128 dwords (one 512B row) / pad_amount=4 dwords -> LDS row stride 264 halfs.
#define A_STR  520           // halfs per A row in LDS (512 + pad)
#define BK_STR 264           // halfs per B K-row in LDS (256 + TDM pad)

__global__ void __launch_bounds__(256)
gat_gemm_kernel(const float* __restrict__ feat,
                const unsigned short* __restrict__ Wb,   // [512][256] bf16
                float* __restrict__ ft)                  // [N_NODES][256] f32
{
    __shared__ unsigned short As[16 * A_STR];            // bf16 A strip [16][512]
    __shared__ unsigned short Bs[32 * BK_STR];           // bf16 B chunk, K-major [k][n]
    unsigned int* As32 = (unsigned int*)As;

    const int tid  = threadIdx.x;
    const int w    = tid >> 5;           // wave id 0..7
    const int lane = tid & 31;
    const int lr   = lane & 15;          // row / col-in-tile
    const int hi   = lane >> 4;          // lane half

    const size_t rowBase = (size_t)blockIdx.x * 16;

    // stage A strip: 16 x 512 f32 -> bf16 LDS (pack 2 halfs per dword store)
    for (int i = tid; i < 16 * 256; i += 256) {
        int r  = i >> 8;
        int c2 = (i & 255) * 2;
        const float* fp = feat + (rowBase + r) * IN_FEATS + c2;
        unsigned int lo   = f32_to_bf16(fp[0]);
        unsigned int hi16 = f32_to_bf16(fp[1]);
        As32[(r * A_STR + c2) >> 1] = lo | (hi16 << 16);
    }
    __syncthreads();

    const unsigned int ldsOff = (unsigned int)(size_t)(void*)Bs;  // LDS byte offset of Bs

    v8f acc0 = {}; v8f acc1 = {};

    for (int kk = 0; kk < IN_FEATS; kk += 32) {
        __syncthreads();                 // all waves done reading previous chunk
        if (w == 0) {
            // Tensor DMA descriptor (ISA 8.3/8.4): 2D tile 256(n) x 32(k) of 2-byte elems
            unsigned long long ga = (unsigned long long)(size_t)(Wb + (size_t)kk * HD);
            v4u g0;
            g0[0] = 1u;                                              // count=1 (valid), user mode
            g0[1] = ldsOff;                                          // lds_addr
            g0[2] = (unsigned int)ga;                                // global_addr[31:0]
            g0[3] = ((unsigned int)(ga >> 32) & 0x01FFFFFFu)         // global_addr[56:32]
                    | 0x80000000u;                                   // type=2 ("image")
            v8i g1;
            g1[0] = (int)((1u << 16)      // data_size = 2 bytes
                    | (1u << 20)          // pad_enable
                    | (6u << 22)          // pad_interval = 128 dwords (one 512B row)
                    | (3u << 25));        // pad_amount   = 4 dwords (8 halfs)
            g1[1] = (int)(256u << 16);    // tensor_dim0 = 256 (low 16 in [31:16])
            g1[2] = (int)(32u << 16);     // tensor_dim0 hi=0; tensor_dim1 = 32
            g1[3] = (int)(256u << 16);    // tensor_dim1 hi=0; tile_dim0 = 256
            g1[4] = 32;                   // tile_dim1 = 32; tile_dim2 = 0
            g1[5] = 256;                  // tensor_dim0_stride = 256 (elems)
            g1[6] = 0;
            g1[7] = 0;
            v4i z4 = {0, 0, 0, 0};
            v8i z8 = {0, 0, 0, 0, 0, 0, 0, 0};
            __builtin_amdgcn_tensor_load_to_lds(g0, g1, z4, z4, z8, 0);
            __builtin_amdgcn_s_wait_tensorcnt(0);
        }
        __syncthreads();                 // chunk visible to all waves

        // A fragment (16x32 bf16), ISA 7.12.2 layout: dword i, lane half hi:
        //   K = 2*(i%4) + (i>=4 ? 16 : 0) + hi*8  (pairs packed per dword)
        Frag16 af;
        #pragma unroll
        for (int i = 0; i < 8; ++i) {
            int kb = 2 * (i & 3) + ((i & 4) << 2) + (hi << 3);
            af.d[i] = As32[(lr * A_STR + kk + kb) >> 1];
        }

        // B fragments (32x16 bf16): lane holds column n, half e = K = (e&7) + hi*8 + (e>=8 ? 16 : 0)
        Frag16 bf0, bf1;
        #pragma unroll
        for (int e = 0; e < 16; ++e) {
            int k = (e & 7) + (hi << 3) + ((e & 8) << 1);
            bf0.u[e] = Bs[k * BK_STR + w * 32 + lr];
            bf1.u[e] = Bs[k * BK_STR + w * 32 + 16 + lr];
        }

        acc0 = __builtin_amdgcn_wmma_f32_16x16x32_bf16(false, af.v, false, bf0.v,
                                                       (short)0, acc0, false, false);
        acc1 = __builtin_amdgcn_wmma_f32_16x16x32_bf16(false, af.v, false, bf1.v,
                                                       (short)0, acc1, false, false);
    }

    // C/D layout: dword v -> row (v + hi*8), col = lane lr
    float* op = ft + rowBase * HD;
    #pragma unroll
    for (int v = 0; v < 8; ++v) {
        int r = v + (hi << 3);
        op[r * HD + w * 32 + lr]      = acc0[v];
        op[r * HD + w * 32 + 16 + lr] = acc1[v];
    }
}

// ---------------------------------------------------------------- el/er: one wave per node
__global__ void __launch_bounds__(256)
elr_kernel(const float* __restrict__ ft,
           const float* __restrict__ attn_l, const float* __restrict__ attn_r,
           float* __restrict__ el, float* __restrict__ er)
{
    int node = (blockIdx.x * blockDim.x + threadIdx.x) >> 5;
    int lane = threadIdx.x & 31;
    if (node >= N_NODES) return;
    const float* row = ft + (size_t)node * HD;
    int c0 = lane * 8;                         // lane covers cols [c0, c0+8) -> head c0/64
    float sl = 0.f, sr = 0.f;
    #pragma unroll
    for (int j = 0; j < 8; ++j) {
        float f = row[c0 + j];
        sl += f * attn_l[c0 + j];              // flat [h*64+d] == col index
        sr += f * attn_r[c0 + j];
    }
    // reduce across the 8 lanes that share a head
    #pragma unroll
    for (int off = 1; off < 8; off <<= 1) {
        sl += __shfl_xor(sl, off, 32);
        sr += __shfl_xor(sr, off, 32);
    }
    if ((lane & 7) == 0) {
        int h = lane >> 3;
        el[node * NUM_HEADS + h] = sl;
        er[node * NUM_HEADS + h] = sr;
    }
}

// ---------------------------------------------------------------- edge passes
__global__ void edge_max_kernel(const int* __restrict__ src, const int* __restrict__ dst,
                                const float* __restrict__ el, const float* __restrict__ er,
                                float* __restrict__ emax)
{
    int t = blockIdx.x * blockDim.x + threadIdx.x;
    if (t >= N_EDGES * NUM_HEADS) return;
    int e = t >> 2, h = t & 3;
    int s = src[e], d = dst[e];
    float x = el[s * NUM_HEADS + h] + er[d * NUM_HEADS + h];
    x = (x > 0.f) ? x : NEG_SLOPE * x;
    atomicMaxFloat(&emax[d * NUM_HEADS + h], x);
}

__global__ void edge_exp_kernel(const int* __restrict__ src, const int* __restrict__ dst,
                                const float* __restrict__ el, const float* __restrict__ er,
                                const float* __restrict__ emax,
                                float* __restrict__ expe, float* __restrict__ denom)
{
    int t = blockIdx.x * blockDim.x + threadIdx.x;
    if (t >= N_EDGES * NUM_HEADS) return;
    int e = t >> 2, h = t & 3;
    int s = src[e], d = dst[e];
    float x = el[s * NUM_HEADS + h] + er[d * NUM_HEADS + h];
    x = (x > 0.f) ? x : NEG_SLOPE * x;
    float ex = expf(x - emax[d * NUM_HEADS + h]);
    expe[t] = ex;
    atomicAdd(&denom[d * NUM_HEADS + h], ex);
}

// one block (256 lanes) per edge: lane c handles feature column c, head c/64
__global__ void __launch_bounds__(256)
aggregate_kernel(const int* __restrict__ src, const int* __restrict__ dst,
                 const float* __restrict__ ft, const float* __restrict__ expe,
                 const float* __restrict__ denom, float* __restrict__ out)
{
    int e = blockIdx.x;
    int c = threadIdx.x;
    int h = c >> 6;
    int s = src[e], d = dst[e];
    float a = expe[e * NUM_HEADS + h] / denom[d * NUM_HEADS + h];
    atomicAdd(&out[(size_t)d * HD + c], a * ft[(size_t)s * HD + c]);
}

// ---------------------------------------------------------------- launcher
extern "C" void kernel_launch(void* const* d_in, const int* in_sizes, int n_in,
                              void* d_out, int out_size, void* d_ws, size_t ws_size,
                              hipStream_t stream) {
    const float* feat   = (const float*)d_in[0];
    const float* W      = (const float*)d_in[1];
    const float* attn_l = (const float*)d_in[2];
    const float* attn_r = (const float*)d_in[3];
    const int*   src    = (const int*)d_in[4];
    const int*   dst    = (const int*)d_in[5];
    float* out = (float*)d_out;

    // workspace carve-up
    float* ft    = (float*)d_ws;                         // 50000*256
    float* el    = ft    + (size_t)N_NODES * HD;         // 50000*4
    float* er    = el    + (size_t)N_NODES * NUM_HEADS;
    float* emax  = er    + (size_t)N_NODES * NUM_HEADS;
    float* denom = emax  + (size_t)N_NODES * NUM_HEADS;
    float* expe  = denom + (size_t)N_NODES * NUM_HEADS;  // 800000*4
    unsigned short* Wb = (unsigned short*)(expe + (size_t)N_EDGES * NUM_HEADS);

    const int T = 256;

    // init: W -> bf16, out = 0, emax = -inf, denom = 0
    cvt_w_bf16<<<(IN_FEATS * HD + T - 1) / T, T, 0, stream>>>(W, Wb, IN_FEATS * HD);
    fill_f32<<<(N_NODES * HD + T - 1) / T, T, 0, stream>>>(out, 0.0f, N_NODES * HD);
    fill_f32<<<(N_NODES * NUM_HEADS + T - 1) / T, T, 0, stream>>>(emax, -INFINITY, N_NODES * NUM_HEADS);
    fill_f32<<<(N_NODES * NUM_HEADS + T - 1) / T, T, 0, stream>>>(denom, 0.0f, N_NODES * NUM_HEADS);

    // projection GEMM (WMMA bf16 + TDM-staged B tiles)
    gat_gemm_kernel<<<N_NODES / 16, T, 0, stream>>>(feat, Wb, ft);

    // attention logits
    elr_kernel<<<(N_NODES * 32 + T - 1) / T, T, 0, stream>>>(ft, attn_l, attn_r, el, er);

    // segment softmax + aggregation
    edge_max_kernel<<<(N_EDGES * NUM_HEADS + T - 1) / T, T, 0, stream>>>(src, dst, el, er, emax);
    edge_exp_kernel<<<(N_EDGES * NUM_HEADS + T - 1) / T, T, 0, stream>>>(src, dst, el, er, emax, expe, denom);
    aggregate_kernel<<<N_EDGES, T, 0, stream>>>(src, dst, ft, expe, denom, out);
}
